// Match_C_36232344109133
// MI455X (gfx1250) — matrix operand order repaired and checked by reading, not compiled, not run on hardware
//
#include <hip/hip_runtime.h>

typedef __attribute__((ext_vector_type(16))) _Float16 v16h;
typedef __attribute__((ext_vector_type(8)))  _Float16 v8h;
typedef __attribute__((ext_vector_type(8)))  float    v8f;
typedef unsigned int u32x4 __attribute__((ext_vector_type(4)));
typedef int          i32x4 __attribute__((ext_vector_type(4)));
typedef int          i32x8 __attribute__((ext_vector_type(8)));

#define IN_DIM   128
#define H_DIM    256
#define MROWS    8192
#define NROWS    8192
#define TOPK     512
#define CAP      8192   // candidate buffer entries

#if defined(__has_builtin)
#if __has_builtin(__builtin_amdgcn_tensor_load_to_lds) && \
    __has_builtin(__builtin_amdgcn_s_wait_tensorcnt)
#define USE_TDM 1
#endif
#endif

union F16Frag { v16h v; v8h g[2]; _Float16 h[16]; };

__device__ __forceinline__ unsigned mono_key(float f) {
    unsigned u = __float_as_uint(f);
    return (u & 0x80000000u) ? ~u : (u | 0x80000000u);
}

#ifdef USE_TDM
// Tensor Data Mover: DMA a [rows x cols] f16 tile (row stride `strideElems`)
// from global memory into LDS at byte offset `ldsByteAddr`. ISA ch.8 D# layout.
// This toolchain exposes the 6-arg builtin: (g0, g1, g2, g3, g4, cpol).
__device__ __forceinline__ void tdm_load_2d_f16(
    unsigned ldsByteAddr, const _Float16* gptr,
    unsigned rows, unsigned cols,
    unsigned tensorW, unsigned tensorH, unsigned strideElems) {
    unsigned long long ga = (unsigned long long)(__UINTPTR_TYPE__)gptr;
    u32x4 g0;
    g0[0] = 0x1u;                                   // count=1, user descriptor
    g0[1] = ldsByteAddr;                            // lds_addr
    g0[2] = (unsigned)(ga & 0xFFFFFFFFu);           // global_addr[31:0]
    g0[3] = (unsigned)((ga >> 32) & 0x1FFFFFFu) | (2u << 30);  // addr[56:32], type=2
    i32x8 g1;
    g1[0] = (int)(1u << 16);                        // data_size=1 (2 bytes)
    g1[1] = (int)((tensorW & 0xFFFFu) << 16);       // tensor_dim0 lo16
    g1[2] = (int)((tensorW >> 16) | ((tensorH & 0xFFFFu) << 16)); // dim0 hi / dim1 lo
    g1[3] = (int)((tensorH >> 16) | (cols << 16));  // dim1 hi / tile_dim0
    g1[4] = (int)(rows & 0xFFFFu);                  // tile_dim1 (tile_dim2=0)
    g1[5] = (int)strideElems;                       // tensor_dim0_stride lo32
    g1[6] = 0;
    g1[7] = 0;
    i32x4 z4 = {0, 0, 0, 0};
    i32x8 z8 = {0, 0, 0, 0, 0, 0, 0, 0};
    __builtin_amdgcn_tensor_load_to_lds(g0, g1, z4, z4, z8, 0);
}
#endif

// ---------------------------------------------------------------------------
// Init scratch state (harness poisons ws once; we must re-init every call).
// ---------------------------------------------------------------------------
__global__ void init_kernel(unsigned* hist, unsigned* state, unsigned* candCount) {
    int t = threadIdx.x;
    if (t < 256) hist[t] = 0;
    if (t == 0) {
        state[0] = 0;      // prefix
        state[1] = 0;      // mask
        state[2] = TOPK;   // need
        state[3] = 0;      // final threshold key
        state[4] = 0;      // take-equal count
        *candCount = 0;
    }
}

// ---------------------------------------------------------------------------
// One-time weight prep: W [K,N] f32 row-major -> Wt [N,K] f16 row-major.
// Makes every WMMA B-fragment a single contiguous 32-byte load.
// ---------------------------------------------------------------------------
__global__ void prep_w_kernel(const float* __restrict__ W,
                              _Float16* __restrict__ Wt, int K, int N) {
    int idx = blockIdx.x * blockDim.x + threadIdx.x;
    if (idx >= K * N) return;
    int n = idx / K, k = idx - n * K;
    Wt[idx] = (_Float16)W[(size_t)k * N + n];
}

// ---------------------------------------------------------------------------
// MLP: out = (relu(X @ Wa + ba)) @ Wb + bb, stored as f16 [rows, 256].
// Weights pre-transposed to f16 [N,K]; one wave per 16-row tile; hidden
// activations staged per-wave in LDS.
// ---------------------------------------------------------------------------
__global__ void __launch_bounds__(256) mlp_kernel(
    const float* __restrict__ X, const _Float16* __restrict__ Wta,
    const float* __restrict__ ba, const _Float16* __restrict__ Wtb,
    const float* __restrict__ bb, _Float16* __restrict__ out) {

    __shared__ __align__(32) _Float16 hbuf[8][16 * H_DIM];

    const int tid    = threadIdx.x;
    const int w      = tid >> 5;
    const int lane   = tid & 31;
    const int halfId = lane >> 4;
    const int r      = lane & 15;      // A-row / B-col / D-col within tile
    const int r0     = blockIdx.x * 128 + w * 16;
    _Float16* hrow = &hbuf[w][0];

    // ---- layer 1: h = relu(X @ Wa + ba), X is [rows,128] f32 ----
    for (int ntb = 0; ntb < 16; ntb += 4) {
        v8f acc[4] = {};
        for (int kc = 0; kc < 4; ++kc) {        // K = 128 in chunks of 32
            const int kbase = kc * 32;
            F16Frag a;
            v8f x0 = *(const v8f*)&X[(size_t)(r0 + r) * IN_DIM + kbase + halfId * 16];
            v8f x1 = *(const v8f*)&X[(size_t)(r0 + r) * IN_DIM + kbase + halfId * 16 + 8];
#pragma unroll
            for (int j = 0; j < 8; ++j) {
                a.h[j]     = (_Float16)x0[j];
                a.h[8 + j] = (_Float16)x1[j];
            }
#pragma unroll
            for (int nt = 0; nt < 4; ++nt) {
                const int n0 = (ntb + nt) * 16;
                F16Frag b;
                b.v = *(const v16h*)&Wta[(size_t)(n0 + r) * IN_DIM + kbase + halfId * 16];
                acc[nt] = __builtin_amdgcn_wmma_f32_16x16x32_f16(
                    false, a.v, false, b.v, (short)0, acc[nt], false, false);
            }
        }
#pragma unroll
        for (int nt = 0; nt < 4; ++nt) {
            const int n0 = (ntb + nt) * 16;
            const float bias = ba[n0 + r];
#pragma unroll
            for (int g = 0; g < 8; ++g) {
                float v = acc[nt][g] + bias;
                v = v > 0.0f ? v : 0.0f;
                hrow[(g + halfId * 8) * H_DIM + n0 + r] = (_Float16)v;
            }
        }
    }

    // ---- layer 2: out = h @ Wb + bb, h is per-wave in LDS (f16) ----
    for (int ntb = 0; ntb < 16; ntb += 4) {
        v8f acc[4] = {};
        for (int kc = 0; kc < 8; ++kc) {        // K = 256 in chunks of 32
            const int kbase = kc * 32;
            F16Frag a;
            const _Float16* arow = &hrow[r * H_DIM + kbase];
            a.g[0] = *(const v8h*)&arow[halfId * 16];
            a.g[1] = *(const v8h*)&arow[halfId * 16 + 8];
#pragma unroll
            for (int nt = 0; nt < 4; ++nt) {
                const int n0 = (ntb + nt) * 16;
                F16Frag b;
                b.v = *(const v16h*)&Wtb[(size_t)(n0 + r) * H_DIM + kbase + halfId * 16];
                acc[nt] = __builtin_amdgcn_wmma_f32_16x16x32_f16(
                    false, a.v, false, b.v, (short)0, acc[nt], false, false);
            }
        }
#pragma unroll
        for (int nt = 0; nt < 4; ++nt) {
            const int n0 = (ntb + nt) * 16;
            const float bias = bb[n0 + r];
#pragma unroll
            for (int g = 0; g < 8; ++g)
                out[(size_t)(r0 + g + halfId * 8) * H_DIM + n0 + r] =
                    (_Float16)(acc[nt][g] + bias);
        }
    }
}

// ---------------------------------------------------------------------------
// sim = A @ B^T, A/B f16 [8192,256], C f32 [8192,8192].
// 128x128 tile per 256-thread block. Entire 128x256 A and B panels staged in
// LDS (128 KB) by the Tensor Data Mover (one descriptor each, TENSORcnt),
// then the K loop is pure ds_load_b128 + v_wmma with no barriers.
// Waves arranged 4x2: each wave owns 32 rows x 64 cols (2 A-frags x 4 B-frags
// per WMMA octet -> 24 B/lane of LDS per WMMA). Fused level-0 radix histogram.
// ---------------------------------------------------------------------------
__global__ void __launch_bounds__(256) gemm_kernel(
    const _Float16* __restrict__ A, const _Float16* __restrict__ B,
    float* __restrict__ sim, unsigned* __restrict__ hist) {

    __shared__ __align__(32) _Float16 as[128 * H_DIM];   // 64 KB
    __shared__ __align__(32) _Float16 bs[128 * H_DIM];   // 64 KB
    __shared__ unsigned lhist[256];

    const int tid    = threadIdx.x;
    const int w      = tid >> 5;
    const int wq     = w >> 1;          // row group 0..3 (32 rows each)
    const int wn     = w & 1;           // col half 0..1  (64 cols each)
    const int lane   = tid & 31;
    const int halfId = lane >> 4;
    const int r      = lane & 15;
    const int m0blk  = blockIdx.y * 128;
    const int n0     = blockIdx.x * 128;

    lhist[tid] = 0;

#ifdef USE_TDM
    if (w == 0) {
        tdm_load_2d_f16((unsigned)(__UINTPTR_TYPE__)as,
                        A + (size_t)m0blk * H_DIM, 128, H_DIM, H_DIM, MROWS, H_DIM);
        tdm_load_2d_f16((unsigned)(__UINTPTR_TYPE__)bs,
                        B + (size_t)n0 * H_DIM, 128, H_DIM, H_DIM, NROWS, H_DIM);
        __builtin_amdgcn_s_wait_tensorcnt(0);
    }
    __syncthreads();
#else
    {   // cooperative panel staging fallback
        const int row = tid >> 1;
        const int h0  = (tid & 1) * 128;
#pragma unroll
        for (int i = 0; i < 8; ++i) {
            *(v16h*)&as[row * H_DIM + h0 + i * 16] =
                *(const v16h*)&A[(size_t)(m0blk + row) * H_DIM + h0 + i * 16];
            *(v16h*)&bs[row * H_DIM + h0 + i * 16] =
                *(const v16h*)&B[(size_t)(n0 + row) * H_DIM + h0 + i * 16];
        }
    }
    __syncthreads();
#endif

    v8f acc[8] = {};    // [mf*4 + nt]
#pragma unroll
    for (int kc = 0; kc < 8; ++kc) {
        const int k0 = kc * 32;
        F16Frag a0, a1;
        const _Float16* ar0 = &as[(wq * 32 + r) * H_DIM + k0];
        const _Float16* ar1 = &as[(wq * 32 + 16 + r) * H_DIM + k0];
        a0.g[0] = *(const v8h*)&ar0[halfId * 16];
        a0.g[1] = *(const v8h*)&ar0[halfId * 16 + 8];
        a1.g[0] = *(const v8h*)&ar1[halfId * 16];
        a1.g[1] = *(const v8h*)&ar1[halfId * 16 + 8];
#pragma unroll
        for (int nt = 0; nt < 4; ++nt) {
            F16Frag b;
            b.v = *(const v16h*)&bs[(wn * 64 + nt * 16 + r) * H_DIM + k0 + halfId * 16];
            acc[nt]     = __builtin_amdgcn_wmma_f32_16x16x32_f16(
                false, a0.v, false, b.v, (short)0, acc[nt],     false, false);
            acc[4 + nt] = __builtin_amdgcn_wmma_f32_16x16x32_f16(
                false, a1.v, false, b.v, (short)0, acc[4 + nt], false, false);
        }
    }

    // epilogue: store sim (non-temporal, it exceeds L2) + byte-3 key histogram
#pragma unroll
    for (int mf = 0; mf < 2; ++mf) {
#pragma unroll
        for (int nt = 0; nt < 4; ++nt) {
#pragma unroll
            for (int g = 0; g < 8; ++g) {
                float v = acc[mf * 4 + nt][g];
                const int row = m0blk + wq * 32 + mf * 16 + g + halfId * 8;
                const int col = n0 + wn * 64 + nt * 16 + r;
                __builtin_nontemporal_store(v, &sim[(size_t)row * NROWS + col]);
                atomicAdd(&lhist[mono_key(v) >> 24], 1u);
            }
        }
    }
    __syncthreads();
    if (lhist[tid]) atomicAdd(&hist[tid], lhist[tid]);
}

// ---------------------------------------------------------------------------
// Radix-select support: serial 256-bin scan (tiny), per-byte histogram pass,
// candidate collection, and final bitonic sort of (key<<32)|~index.
// ---------------------------------------------------------------------------
__global__ void scan_kernel(unsigned* hist, unsigned* state, unsigned* candCount,
                            int shift, int isLast) {
    if (threadIdx.x != 0 || blockIdx.x != 0) return;
    unsigned need = state[2];
    unsigned cum = 0;
    int b = 255;
    for (; b > 0; --b) {
        if (cum + hist[b] >= need) break;
        cum += hist[b];
    }
    state[0] |= ((unsigned)b) << shift;
    state[1] |= 0xFFu << shift;
    state[2] = need - cum;
    for (int i = 0; i < 256; ++i) hist[i] = 0;
    if (isLast) {
        state[3] = state[0];
        state[4] = state[2];
        *candCount = 0;
    }
}

__global__ void __launch_bounds__(256) hist_kernel(
    const float* __restrict__ sim, const unsigned* __restrict__ state,
    unsigned* __restrict__ hist, int shift, long long total) {
    __shared__ unsigned lhist[256];
    const int tid = threadIdx.x;
    lhist[tid] = 0;
    __syncthreads();
    const unsigned mask = state[1], prefix = state[0];
    long long i = (long long)blockIdx.x * blockDim.x + tid;
    const long long stride = (long long)gridDim.x * blockDim.x;
    for (; i < total; i += stride) {
        unsigned key = mono_key(__builtin_nontemporal_load(&sim[i]));
        if ((key & mask) == prefix)
            atomicAdd(&lhist[(key >> shift) & 255u], 1u);
    }
    __syncthreads();
    if (lhist[tid]) atomicAdd(&hist[tid], lhist[tid]);
}

__global__ void __launch_bounds__(256) collect_kernel(
    const float* __restrict__ sim, const unsigned* __restrict__ state,
    unsigned* __restrict__ candCount, unsigned long long* __restrict__ cand,
    long long total) {
    const unsigned T = state[3];
    long long i = (long long)blockIdx.x * blockDim.x + threadIdx.x;
    const long long stride = (long long)gridDim.x * blockDim.x;
    for (; i < total; i += stride) {
        unsigned key = mono_key(__builtin_nontemporal_load(&sim[i]));
        if (key >= T) {
            unsigned pos = atomicAdd(candCount, 1u);
            if (pos < CAP)
                cand[pos] = ((unsigned long long)key << 32) |
                            (unsigned)(~(unsigned)i);
        }
    }
}

__global__ void __launch_bounds__(1024) final_kernel(
    const unsigned long long* __restrict__ cand,
    const unsigned* __restrict__ candCount, float* __restrict__ outIdx) {
    __shared__ unsigned long long arr[CAP];
    const int tid = threadIdx.x;
    const int nth = blockDim.x;
    unsigned n = *candCount;
    if (n > CAP) n = CAP;
    for (int i = tid; i < CAP; i += nth)
        arr[i] = (i < (int)n) ? cand[i] : 0ull;
    __syncthreads();
    // bitonic sort, descending: (key desc, index asc via ~index in low bits)
    for (int k = 2; k <= CAP; k <<= 1) {
        for (int j = k >> 1; j > 0; j >>= 1) {
            for (int i = tid; i < CAP; i += nth) {
                int ixj = i ^ j;
                if (ixj > i) {
                    unsigned long long x = arr[i], y = arr[ixj];
                    bool up = ((i & k) == 0);
                    if (up ? (x < y) : (x > y)) { arr[i] = y; arr[ixj] = x; }
                }
            }
            __syncthreads();
        }
    }
    for (int t = tid; t < TOPK; t += nth) {
        unsigned idx = ~(unsigned)(arr[t] & 0xFFFFFFFFull);
        outIdx[2 * t + 0] = (float)(idx / (unsigned)NROWS);
        outIdx[2 * t + 1] = (float)(idx % (unsigned)NROWS);
    }
}

// ---------------------------------------------------------------------------
extern "C" void kernel_launch(void* const* d_in, const int* in_sizes, int n_in,
                              void* d_out, int out_size, void* d_ws, size_t ws_size,
                              hipStream_t stream) {
    (void)in_sizes; (void)n_in; (void)out_size; (void)ws_size;

    const float* desc0 = (const float*)d_in[0];
    const float* desc1 = (const float*)d_in[1];
    const float* W0a = (const float*)d_in[2];
    const float* b0a = (const float*)d_in[3];
    const float* W0b = (const float*)d_in[4];
    const float* b0b = (const float*)d_in[5];
    const float* W1a = (const float*)d_in[6];
    const float* b1a = (const float*)d_in[7];
    const float* W1b = (const float*)d_in[8];
    const float* b1b = (const float*)d_in[9];

    float* out    = (float*)d_out;
    float* outIdx = out;            // [512,2] indices (as f32)
    float* sim    = out + TOPK * 2; // [8192*8192] f32

    // scratch layout
    char* ws = (char*)d_ws;
    const size_t mdBytes = (size_t)MROWS * H_DIM * sizeof(_Float16); // 4 MB
    _Float16* md0 = (_Float16*)ws;
    _Float16* md1 = (_Float16*)(ws + mdBytes);
    unsigned* hist      = (unsigned*)(ws + 2 * mdBytes);
    unsigned* state     = hist + 256;
    unsigned* candCount = state + 16;
    unsigned long long* cand = (unsigned long long*)(ws + 2 * mdBytes + 4096);
    char* wbase = ws + 2 * mdBytes + 4096 + (size_t)CAP * 8;
    _Float16* wt0a = (_Float16*)(wbase);                          // 64 KB
    _Float16* wt0b = (_Float16*)(wbase + (64 << 10));             // 128 KB
    _Float16* wt1a = (_Float16*)(wbase + (192 << 10));            // 64 KB
    _Float16* wt1b = (_Float16*)(wbase + (256 << 10));            // 128 KB

    const long long total = (long long)MROWS * NROWS;

    init_kernel<<<1, 256, 0, stream>>>(hist, state, candCount);
    prep_w_kernel<<<(IN_DIM * H_DIM + 255) / 256, 256, 0, stream>>>(W0a, wt0a, IN_DIM, H_DIM);
    prep_w_kernel<<<(H_DIM * H_DIM + 255) / 256, 256, 0, stream>>>(W0b, wt0b, H_DIM, H_DIM);
    prep_w_kernel<<<(IN_DIM * H_DIM + 255) / 256, 256, 0, stream>>>(W1a, wt1a, IN_DIM, H_DIM);
    prep_w_kernel<<<(H_DIM * H_DIM + 255) / 256, 256, 0, stream>>>(W1b, wt1b, H_DIM, H_DIM);

    mlp_kernel<<<MROWS / 128, 256, 0, stream>>>(desc0, wt0a, b0a, wt0b, b0b, md0);
    mlp_kernel<<<NROWS / 128, 256, 0, stream>>>(desc1, wt1a, b1a, wt1b, b1b, md1);

    dim3 gg(NROWS / 128, MROWS / 128);
    gemm_kernel<<<gg, 256, 0, stream>>>(md0, md1, sim, hist);

    scan_kernel<<<1, 1, 0, stream>>>(hist, state, candCount, 24, 0);
    hist_kernel<<<2048, 256, 0, stream>>>(sim, state, hist, 16, total);
    scan_kernel<<<1, 1, 0, stream>>>(hist, state, candCount, 16, 0);
    hist_kernel<<<2048, 256, 0, stream>>>(sim, state, hist, 8, total);
    scan_kernel<<<1, 1, 0, stream>>>(hist, state, candCount, 8, 0);
    hist_kernel<<<2048, 256, 0, stream>>>(sim, state, hist, 0, total);
    scan_kernel<<<1, 1, 0, stream>>>(hist, state, candCount, 0, 1);

    collect_kernel<<<2048, 256, 0, stream>>>(sim, state, candCount, cand, total);
    final_kernel<<<1, 1024, 0, stream>>>(cand, candCount, outIdx);
}
